// ABMIL_82532091559957
// MI455X (gfx1250) — matrix operand order repaired and checked by reading, not compiled, not run on hardware
//
#include <hip/hip_runtime.h>
#include <hip/hip_bf16.h>
#include <math.h>

// ---------------------------------------------------------------------------
// ABMIL gated-attention pooling for MI455X (gfx1250, wave32, WMMA).
// B=16, N=4096, D=1024, H=256.
// ---------------------------------------------------------------------------

typedef __attribute__((ext_vector_type(8)))  float  v8f;
typedef __attribute__((ext_vector_type(8)))  __bf16 v8bf;
typedef __attribute__((ext_vector_type(16))) __bf16 bf16x16;

#define AB_B 16
#define AB_N 4096
#define AB_D 1024
#define AB_H 256
#define AB_M (AB_B * AB_N)          // 65536 rows
#define LDS_STRIDE 136              // 128 K-chunk + 8 pad (bf16 elems)

__device__ __forceinline__ unsigned int bf16_rne(float f) {
    unsigned int u = __builtin_bit_cast(unsigned int, f);
    u += 0x7fffu + ((u >> 16) & 1u);
    return u >> 16;
}
__device__ __forceinline__ unsigned int pack2_bf16(float lo, float hi) {
    return bf16_rne(lo) | (bf16_rne(hi) << 16);
}

// ---------------------------------------------------------------------------
// Kernel 1: convert V_w / U_w fp32 [D][H] -> bf16 transposed [H][D].
// ---------------------------------------------------------------------------
__global__ __launch_bounds__(256) void abmil_cvtw_kernel(
    const float* __restrict__ Vw, const float* __restrict__ Uw,
    unsigned short* __restrict__ wV, unsigned short* __restrict__ wU)
{
    int t = blockIdx.x * 256 + threadIdx.x;      // 0 .. 2*262144-1
    int which = t >> 18;
    int e = t & 262143;
    int d = e & (AB_D - 1);
    int h = e >> 10;
    const float* src = which ? Uw : Vw;
    unsigned short* dst = which ? wU : wV;
    dst[h * AB_D + d] = (unsigned short)bf16_rne(src[d * AB_H + h]);
}

// ---------------------------------------------------------------------------
// B-fragment group load: two 16-col tiles x {V,U}. kg already includes the
// per-lane K-half offset (lanes 0-15: K 0..15, lanes 16-31: K 16..31).
// ---------------------------------------------------------------------------
__device__ __forceinline__ void abmil_loadB(
    const unsigned short* __restrict__ wV,
    const unsigned short* __restrict__ wU,
    int hbase, int hlane, int kg,
    bf16x16& bv0, bf16x16& bu0, bf16x16& bv1, bf16x16& bu1)
{
    const size_t b0 = (size_t)(hbase + hlane) * AB_D + kg;
    const size_t b1 = (size_t)(hbase + 16 + hlane) * AB_D + kg;
    bv0 = *(const bf16x16*)(wV + b0);
    bu0 = *(const bf16x16*)(wU + b0);
    bv1 = *(const bf16x16*)(wV + b1);
    bu1 = *(const bf16x16*)(wU + b1);
}

// ---------------------------------------------------------------------------
// Kernel 2: fused score GEMM.
//   scores[m] = sum_h tanh(x@Vw + Vb)[h] * sigmoid(x@Uw + Ub)[h] * Ww[h] + Wb
// Block = 256 threads (8 waves), owns 32 rows x 256 cols. Wave w owns cols
// [w*32, w*32+32) for BOTH row tiles and BOTH matrices. B fragments are
// prefetched one K-step ahead; x staging is register double-buffered.
// ---------------------------------------------------------------------------
__global__ __launch_bounds__(256) void abmil_scores_kernel(
    const float* __restrict__ x,
    const unsigned short* __restrict__ wV,   // bf16 [H][D]
    const unsigned short* __restrict__ wU,   // bf16 [H][D]
    const float* __restrict__ Vb, const float* __restrict__ Ub,
    const float* __restrict__ Ww, const float* __restrict__ Wb,
    float* __restrict__ scores)
{
    __shared__ unsigned short xs[32 * LDS_STRIDE];   // bf16 tile, 32 x 128 (+pad)
    __shared__ float sred[32][9];                    // [row][wave], padded

    const int tid    = threadIdx.x;
    const int lane   = tid & 31;
    const int wave   = tid >> 5;        // 0..7
    const int hlane  = lane & 15;       // column within 16-wide tile
    const int laneHi = lane >> 4;       // 0 | 1 (K half select)
    const int rowBase = blockIdx.x * 32;
    const int hbase   = wave * 32;      // this wave's 32 columns

    v8f accV[2][2] = {}; v8f accU[2][2] = {};   // [rowTile][colTile]

    // staging mapping: thread -> (row, 16 consecutive K)
    const int srow = tid >> 3;               // 0..31
    const int scol = (tid & 7) * 16;         // 0..112
    const float* xg = x + (size_t)(rowBase + srow) * AB_D + scol;
    unsigned short* xsrow = xs + srow * LDS_STRIDE + scol;

    // A-fragment LDS base offsets for the two row tiles this wave computes.
    const unsigned short* a0base = xs + hlane * LDS_STRIDE;          // rows 0..15
    const unsigned short* a1base = xs + (16 + hlane) * LDS_STRIDE;   // rows 16..31

    // ---- prime the pipeline: first x chunk + first B group ----
    const float4* g0 = (const float4*)xg;
    float4 f0 = g0[0], f1 = g0[1], f2 = g0[2], f3 = g0[3];

    bf16x16 bv0, bu0, bv1, bu1;
    abmil_loadB(wV, wU, hbase, hlane, laneHi * 16, bv0, bu0, bv1, bu1);

    for (int k0 = 0; k0 < AB_D; k0 += 128) {
        __syncthreads();
        // ---- commit staged chunk: fp32 -> bf16 into LDS ----
        uint4 p0, p1;
        p0.x = pack2_bf16(f0.x, f0.y); p0.y = pack2_bf16(f0.z, f0.w);
        p0.z = pack2_bf16(f1.x, f1.y); p0.w = pack2_bf16(f1.z, f1.w);
        p1.x = pack2_bf16(f2.x, f2.y); p1.y = pack2_bf16(f2.z, f2.w);
        p1.z = pack2_bf16(f3.x, f3.y); p1.w = pack2_bf16(f3.z, f3.w);
        ((uint4*)xsrow)[0] = p0;
        ((uint4*)xsrow)[1] = p1;
        __syncthreads();

        // ---- prefetch next chunk's x (wraps on last iter; harmless) ----
        {
            const float4* gn = (const float4*)(xg + ((k0 + 128) & (AB_D - 1)));
            f0 = gn[0]; f1 = gn[1]; f2 = gn[2]; f3 = gn[3];
        }

        // ---- 4 WMMA K-steps of 32, B prefetched one step ahead ----
        #pragma unroll
        for (int ks = 0; ks < 4; ++ks) {
            const int kc = ks * 32;
            // A fragments (16x32 bf16): contiguous K, matching B —
            // lanes 0-15: K kc..kc+15; lanes 16-31: K kc+16..kc+31
            const unsigned short* ap0 = a0base + kc + laneHi * 16;
            v8bf a0lo = *(const v8bf*)ap0;
            v8bf a0hi = *(const v8bf*)(ap0 + 8);
            bf16x16 a0 = __builtin_shufflevector(a0lo, a0hi,
                0,1,2,3,4,5,6,7,8,9,10,11,12,13,14,15);
            const unsigned short* ap1 = a1base + kc + laneHi * 16;
            v8bf a1lo = *(const v8bf*)ap1;
            v8bf a1hi = *(const v8bf*)(ap1 + 8);
            bf16x16 a1 = __builtin_shufflevector(a1lo, a1hi,
                0,1,2,3,4,5,6,7,8,9,10,11,12,13,14,15);

            // prefetch next K-step's B group (wraps at the very end; harmless)
            const int kgn = ((k0 + kc + 32) & (AB_D - 1)) + laneHi * 16;
            bf16x16 nv0, nu0, nv1, nu1;
            abmil_loadB(wV, wU, hbase, hlane, kgn, nv0, nu0, nv1, nu1);

            // 8 WMMAs with the current (already-resident) B group
            accV[0][0] = __builtin_amdgcn_wmma_f32_16x16x32_bf16(
                false, a0, false, bv0, (short)0, accV[0][0], false, false);
            accV[1][0] = __builtin_amdgcn_wmma_f32_16x16x32_bf16(
                false, a1, false, bv0, (short)0, accV[1][0], false, false);
            accU[0][0] = __builtin_amdgcn_wmma_f32_16x16x32_bf16(
                false, a0, false, bu0, (short)0, accU[0][0], false, false);
            accU[1][0] = __builtin_amdgcn_wmma_f32_16x16x32_bf16(
                false, a1, false, bu0, (short)0, accU[1][0], false, false);
            accV[0][1] = __builtin_amdgcn_wmma_f32_16x16x32_bf16(
                false, a0, false, bv1, (short)0, accV[0][1], false, false);
            accV[1][1] = __builtin_amdgcn_wmma_f32_16x16x32_bf16(
                false, a1, false, bv1, (short)0, accV[1][1], false, false);
            accU[0][1] = __builtin_amdgcn_wmma_f32_16x16x32_bf16(
                false, a0, false, bu1, (short)0, accU[0][1], false, false);
            accU[1][1] = __builtin_amdgcn_wmma_f32_16x16x32_bf16(
                false, a1, false, bu1, (short)0, accU[1][1], false, false);

            bv0 = nv0; bu0 = nu0; bv1 = nv1; bu1 = nu1;
        }
    }

    // ---- epilogue: gate, weight by Ww, reduce over H ----
    // C layout: lanes 0-15 -> col=lane, rows i=0..7; lanes 16-31 -> rows 8..15.
    float rowSum[2][8] = {};
    #pragma unroll
    for (int ct = 0; ct < 2; ++ct) {
        const int h = hbase + ct * 16 + hlane;
        const float vb = Vb[h], ub = Ub[h], ww = Ww[h];
        #pragma unroll
        for (int rt = 0; rt < 2; ++rt) {
            #pragma unroll
            for (int i = 0; i < 8; ++i) {
                float tv = tanhf(accV[rt][ct][i] + vb);
                float sg = 1.0f / (1.0f + __expf(-(accU[rt][ct][i] + ub)));
                rowSum[rt][i] += tv * sg * ww;
            }
        }
    }
    #pragma unroll
    for (int rt = 0; rt < 2; ++rt) {
        #pragma unroll
        for (int i = 0; i < 8; ++i) {
            float s = rowSum[rt][i];
            s += __shfl_xor(s, 1);
            s += __shfl_xor(s, 2);
            s += __shfl_xor(s, 4);
            s += __shfl_xor(s, 8);
            rowSum[rt][i] = s;
        }
    }
    if (hlane == 0) {
        #pragma unroll
        for (int rt = 0; rt < 2; ++rt)
            #pragma unroll
            for (int i = 0; i < 8; ++i)
                sred[rt * 16 + laneHi * 8 + i][wave] = rowSum[rt][i];
    }
    __syncthreads();
    if (tid < 32) {
        float s = Wb[0];
        #pragma unroll
        for (int w = 0; w < 8; ++w) s += sred[tid][w];
        scores[rowBase + tid] = s;
    }
}

// ---------------------------------------------------------------------------
// Kernel 3: masked softmax per bag.
// ---------------------------------------------------------------------------
__global__ __launch_bounds__(256) void abmil_softmax_kernel(
    const float* __restrict__ scores, const int* __restrict__ lengths,
    float* __restrict__ attn)
{
    __shared__ float red[256];
    const int b = blockIdx.x;
    const int tid = threadIdx.x;
    int len = lengths[b]; if (len < 1) len = 1;
    const float* s = scores + b * AB_N;
    float* a = attn + b * AB_N;

    float m = -3.0e38f;
    for (int n = tid; n < AB_N; n += 256)
        if (n < len) m = fmaxf(m, s[n]);
    red[tid] = m; __syncthreads();
    for (int off = 128; off > 0; off >>= 1) {
        if (tid < off) red[tid] = fmaxf(red[tid], red[tid + off]);
        __syncthreads();
    }
    m = red[0]; __syncthreads();

    float sum = 0.f;
    for (int n = tid; n < AB_N; n += 256) {
        float e = 0.f;
        if (n < len) e = __expf(s[n] - m);
        a[n] = e;
        sum += e;
    }
    red[tid] = sum; __syncthreads();
    for (int off = 128; off > 0; off >>= 1) {
        if (tid < off) red[tid] += red[tid + off];
        __syncthreads();
    }
    const float inv = 1.0f / red[0];
    for (int n = tid; n < AB_N; n += 256) a[n] *= inv;
}

// ---------------------------------------------------------------------------
// Kernel 4: Z partials. grid (B, D/256, 8 N-chunks), block 256.
// ---------------------------------------------------------------------------
__global__ __launch_bounds__(256) void abmil_zpart_kernel(
    const float* __restrict__ x, const float* __restrict__ attn,
    float* __restrict__ partial)
{
    const int b = blockIdx.x;
    const int d = blockIdx.y * 256 + threadIdx.x;
    const int c = blockIdx.z;
    const int n0 = c * (AB_N / 8);
    const float* xb = x + (size_t)b * AB_N * AB_D + d;
    const float* ab = attn + b * AB_N;
    float acc = 0.f;
    for (int n = n0; n < n0 + AB_N / 8; ++n)
        acc += ab[n] * xb[(size_t)n * AB_D];
    partial[((size_t)c * AB_B + b) * AB_D + d] = acc;
}

// ---------------------------------------------------------------------------
// Kernel 5: reduce partials -> Z [B, D].
// ---------------------------------------------------------------------------
__global__ __launch_bounds__(256) void abmil_zred_kernel(
    const float* __restrict__ partial, float* __restrict__ out)
{
    const int idx = blockIdx.x * 256 + threadIdx.x;     // 0 .. B*D-1
    float s = 0.f;
    #pragma unroll
    for (int c = 0; c < 8; ++c) s += partial[c * (AB_B * AB_D) + idx];
    out[idx] = s;
}

// ---------------------------------------------------------------------------
extern "C" void kernel_launch(void* const* d_in, const int* in_sizes, int n_in,
                              void* d_out, int out_size, void* d_ws, size_t ws_size,
                              hipStream_t stream)
{
    const float* x       = (const float*)d_in[0];
    const int*   lengths = (const int*)d_in[1];
    const float* Vw      = (const float*)d_in[2];
    const float* Vb      = (const float*)d_in[3];
    const float* Uw      = (const float*)d_in[4];
    const float* Ub      = (const float*)d_in[5];
    const float* Ww      = (const float*)d_in[6];
    const float* Wb      = (const float*)d_in[7];
    float* out = (float*)d_out;

    char* ws = (char*)d_ws;
    unsigned short* wV  = (unsigned short*)(ws);             // 512 KB bf16 [H][D]
    unsigned short* wU  = (unsigned short*)(ws + 524288);    // 512 KB
    float* scores       = (float*)(ws + 1048576);            // 256 KB [B*N]
    float* attn         = (float*)(ws + 1310720);            // 256 KB [B*N]
    float* partial      = (float*)(ws + 1572864);            // 512 KB [8][B*D]

    abmil_cvtw_kernel<<<2048, 256, 0, stream>>>(Vw, Uw, wV, wU);
    abmil_scores_kernel<<<AB_M / 32, 256, 0, stream>>>(x, wV, wU, Vb, Ub, Ww, Wb, scores);
    abmil_softmax_kernel<<<AB_B, 256, 0, stream>>>(scores, lengths, attn);
    abmil_zpart_kernel<<<dim3(AB_B, AB_D / 256, 8), 256, 0, stream>>>(x, attn, partial);
    abmil_zred_kernel<<<(AB_B * AB_D) / 256, 256, 0, stream>>>(partial, out);
}
